// Model_XmaxEnergy_LSTMFeatures_87969520157144
// MI455X (gfx1250) — matrix operand order repaired and checked by reading, not compiled, not run on hardware
//
#include <hip/hip_runtime.h>
#include <hip/hip_bf16.h>
#include <math.h>

typedef __attribute__((ext_vector_type(16))) _Float16 v16h;
typedef __attribute__((ext_vector_type(8)))  float    v8f;
typedef __attribute__((ext_vector_type(4)))  unsigned int u32x4;
typedef __attribute__((ext_vector_type(8)))  int      i32x8;
typedef __attribute__((ext_vector_type(4)))  int      i32x4;

#define P_TRACES 65536
#define NEV      4096

__device__ __forceinline__ float sigm(float x) { return 1.0f / (1.0f + __expf(-x)); }

// ---------------- utility ----------------
__global__ void zero_f32(float* p, int n) {
    int i = blockIdx.x * blockDim.x + threadIdx.x;
    if (i < n) p[i] = 0.0f;
}

// ---------------- LSTM input projection ----------------
__global__ void lstm_pre(const float* __restrict__ x, int ldx, int K,
                         const float* __restrict__ Wih, const float* __restrict__ b,
                         float* __restrict__ pre, int P) {
    int idx = blockIdx.x * blockDim.x + threadIdx.x;
    if (idx >= P * 40) return;
    int p = idx / 40, col = idx - p * 40;
    int d = col / 20, j = col - d * 20;
    const float* w  = Wih + (size_t)(d * 20 + j) * K;
    const float* xr = x + (size_t)p * ldx;
    float s = b[d * 20 + j];
    for (int k = 0; k < K; ++k) s += xr[k] * w[k];
    pre[(size_t)p * 40 + col] = s;
}

// ---------------- sequential LSTM scan, one wave per direction ----------------
__global__ void lstm_scan(const float* __restrict__ pre, const float* __restrict__ Whh,
                          float* __restrict__ out, int P) {
    int d    = blockIdx.x;       // 0 = forward, 1 = reverse
    int lane = threadIdx.x;      // wave32
    int j = lane < 20 ? lane : 19;
    const float* wr = Whh + d * 100 + j * 5;
    float w0 = wr[0], w1 = wr[1], w2 = wr[2], w3 = wr[3], w4 = wr[4];
    float h = 0.0f, c = 0.0f;
    int t0 = d ? (P - 1) : 0;
    int dt = d ? -1 : 1;
    for (int n = 0; n < P; ++n) {
        int t = t0 + n * dt;
        float g = pre[(size_t)t * 40 + d * 20 + j];
        g += __shfl(h, 0, 32) * w0 + __shfl(h, 1, 32) * w1 + __shfl(h, 2, 32) * w2 +
             __shfl(h, 3, 32) * w3 + __shfl(h, 4, 32) * w4;
        float ii = __shfl(g, lane, 32);
        float ff = __shfl(g, lane + 5, 32);
        float gg = __shfl(g, lane + 10, 32);
        float oo = __shfl(g, lane + 15, 32);
        float cn = sigm(ff) * c + sigm(ii) * tanhf(gg);
        float hn = sigm(oo) * tanhf(cn);
        if (lane < 5) {
            c = cn; h = hn;
            out[(size_t)t * 10 + d * 5 + lane] = hn;
        }
    }
}

// ---------------- scatter features into (NEV, 6, 20, 22) grid ----------------
__global__ void scatter_grid(const float* __restrict__ seq, const float* __restrict__ Pstart,
                             const int* __restrict__ Xs, const int* __restrict__ Ys,
                             const int* __restrict__ Ev, float* __restrict__ Main, int P) {
    int p = blockIdx.x * blockDim.x + threadIdx.x;
    if (p >= P) return;
    float f = seq[(size_t)p * 10];
    if (f != f) f = -1.0f;
    float ps = Pstart[p];
    if (ps != ps) ps = -1.0f;
    int e = Ev[p], xs = Xs[p], ys = Ys[p];
    size_t base = (size_t)e * 2640 + (size_t)xs * 22 + ys;
    #pragma unroll
    for (int ch = 0; ch < 5; ++ch) Main[base + (size_t)ch * 440] = f;
    Main[base + 5 * 440] = ps;
}

// ---------------- Tensor Data Mover: 1-D contiguous global->LDS (data_size = 8B) ----------------
__device__ __forceinline__ void tdm_load_1d(unsigned lds_byte_addr, const void* gptr, int n8) {
#if __has_builtin(__builtin_amdgcn_tensor_load_to_lds)
    unsigned long long ga = (unsigned long long)gptr;
    u32x4 g0 = { 1u,                                  // count=1 valid descriptor
                 lds_byte_addr,                       // lds_addr
                 (unsigned)(ga & 0xFFFFFFFFu),        // global_addr[31:0]
                 (unsigned)(((ga >> 32) & 0x01FFFFFFu) | (2u << 30)) }; // ga[56:32] | type=2
    i32x8 g1 = { (int)(3u << 16),                     // workgroup_mask=0, data_size=3 (8B)
                 (int)((n8 & 0xFFFF) << 16),          // tensor_dim0[15:0]
                 (int)(((n8 >> 16) & 0xFFFF) | (1 << 16)), // tensor_dim0[31:16] | tensor_dim1=1
                 (int)((n8 & 0xFFFF) << 16),          // tensor_dim1 hi=0 | tile_dim0
                 1,                                   // tile_dim1=1, tile_dim2=0
                 n8,                                  // tensor_dim0_stride[31:0]
                 0, 0 };
    i32x4 z = { 0, 0, 0, 0 };
#if defined(__clang_major__) && (__clang_major__ >= 23)
    i32x8 z8 = { 0, 0, 0, 0, 0, 0, 0, 0 };
    __builtin_amdgcn_tensor_load_to_lds(g0, g1, z, z, z8, 0);
#else
    __builtin_amdgcn_tensor_load_to_lds(g0, g1, z, z, 0);
#endif
    __builtin_amdgcn_s_wait_tensorcnt(0);
#endif
}

// ---------------- conv0: (E,6,20,22) f32 -> (E,32,400) f16, pad H only ----------------
__global__ __launch_bounds__(128)
void conv0_lrelu(const float* __restrict__ inp, const float* __restrict__ w,
                 const float* __restrict__ bias, _Float16* __restrict__ outp) {
    constexpr int CLDS = 8, KTOT = 54, KP = 64, WSRC = 22, PADW = 0;
    constexpr int HP = 22, WP = 22;
    __shared__ _Float16 sIn[CLDS * HP * WP];
    __shared__ _Float16 sW[KP * 32];
    const int e = blockIdx.x, tid = threadIdx.x;

    for (int i = tid; i < CLDS * HP * WP; i += 128) sIn[i] = (_Float16)0.0f;
    __syncthreads();
    const int tot = 6 * 20 * WSRC;
    const float* in = inp + (size_t)e * tot;
    for (int i = tid; i < tot; i += 128) {
        int c = i / (20 * WSRC); int rem = i - c * (20 * WSRC);
        int y = rem / WSRC;      int x  = rem - y * WSRC;
        sIn[c * (HP * WP) + (y + 1) * WP + (x + PADW)] = (_Float16)in[i];
    }
    for (int i = tid; i < KP * 32; i += 128) {
        int k = i >> 5, co = i & 31;
        float v = (k < KTOT) ? w[(size_t)co * KTOT + k] : 0.0f;
        sW[k * 32 + co] = (_Float16)v;
    }
    __syncthreads();

    const int lane = tid & 31, wave = tid >> 5;
    const int hi = lane >> 4, l15 = lane & 15;
    for (int mt = wave; mt < 25; mt += 4) {
        int m = mt * 16 + l15;
        int oy = m / 20, ox = m - oy * 20;
        v8f acc0 = {}, acc1 = {};
        for (int kk = 0; kk < KP; kk += 32) {
            v16h a, b0, b1;
            #pragma unroll
            for (int i2 = 0; i2 < 8; ++i2) {
                int kbA = kk + 2 * i2 + (i2 >= 4 ? 8 : 0) + hi * 8;
                #pragma unroll
                for (int j2 = 0; j2 < 2; ++j2) {
                    int k = kbA + j2;
                    int c = k / 9; int rem = k - c * 9;
                    int r = rem / 3; int s = rem - r * 3;
                    a[2 * i2 + j2] = sIn[c * (HP * WP) + (oy + r) * WP + (ox + s)];
                }
                int kbB = kk + 2 * i2 + (i2 >= 4 ? 8 : 0) + hi * 8;
                b0[2 * i2]     = sW[kbB * 32 + l15];
                b0[2 * i2 + 1] = sW[(kbB + 1) * 32 + l15];
                b1[2 * i2]     = sW[kbB * 32 + 16 + l15];
                b1[2 * i2 + 1] = sW[(kbB + 1) * 32 + 16 + l15];
            }
            acc0 = __builtin_amdgcn_wmma_f32_16x16x32_f16(false, a, false, b0, (short)0, acc0, false, false);
            acc1 = __builtin_amdgcn_wmma_f32_16x16x32_f16(false, a, false, b1, (short)0, acc1, false, false);
        }
        float bv0 = bias[l15], bv1 = bias[16 + l15];
        _Float16* outE = outp + (size_t)e * 12800;
        #pragma unroll
        for (int r2 = 0; r2 < 8; ++r2) {
            int mo = mt * 16 + r2 + hi * 8;
            float v0 = acc0[r2] + bv0; v0 = v0 > 0.0f ? v0 : 0.01f * v0;
            float v1 = acc1[r2] + bv1; v1 = v1 > 0.0f ? v1 : 0.01f * v1;
            outE[(size_t)l15 * 400 + mo]        = (_Float16)v0;
            outE[(size_t)(16 + l15) * 400 + mo] = (_Float16)v1;
        }
    }
}

// ---------------- conv1-3: (E,32,20,20) f16 -> (E,32,400) f16, TDM-staged input ----------------
__global__ __launch_bounds__(128)
void conv32_lrelu(const _Float16* __restrict__ inp, const float* __restrict__ w,
                  const float* __restrict__ bias, _Float16* __restrict__ outp) {
    constexpr int KTOT = 288, KP = 288;
    constexpr int HP = 22, WP = 22;
    __shared__ _Float16 sIn[32 * HP * WP];   // padded input tile
    __shared__ _Float16 sU[12800];           // staging (25600B), later reused as weights (18432B)
    const int e = blockIdx.x, tid = threadIdx.x;
    const int wave = tid >> 5;

    // stage raw input via Tensor Data Mover (wave 0 issues; EXEC ignored by TDM)
    if (wave == 0) {
#if __has_builtin(__builtin_amdgcn_tensor_load_to_lds)
        tdm_load_1d((unsigned)(unsigned long long)(void*)&sU[0],
                    (const void*)(inp + (size_t)e * 12800), 12800 * 2 / 8);
#else
        for (int i = tid; i < 12800; i += 32) sU[i] = inp[(size_t)e * 12800 + i];
#endif
    }
    for (int i = tid; i < 32 * HP * WP; i += 128) sIn[i] = (_Float16)0.0f;
    __syncthreads();
    // re-layout staged (c,y,x) -> padded tile (c, y+1, x+1)
    for (int i = tid; i < 12800; i += 128) {
        int c = i / 400; int rem = i - c * 400;
        int y = rem / 20; int x = rem - y * 20;
        sIn[c * (HP * WP) + (y + 1) * WP + (x + 1)] = sU[i];
    }
    __syncthreads();
    // overwrite staging region with transposed weights [k][cout]
    for (int i = tid; i < KP * 32; i += 128) {
        int k = i >> 5, co = i & 31;
        sU[k * 32 + co] = (_Float16)w[(size_t)co * KTOT + k];
    }
    __syncthreads();

    const int lane = tid & 31;
    const int hi = lane >> 4, l15 = lane & 15;
    for (int mt = wave; mt < 25; mt += 4) {
        int m = mt * 16 + l15;
        int oy = m / 20, ox = m - oy * 20;
        v8f acc0 = {}, acc1 = {};
        for (int kk = 0; kk < KP; kk += 32) {
            v16h a, b0, b1;
            #pragma unroll
            for (int i2 = 0; i2 < 8; ++i2) {
                int kbA = kk + 2 * i2 + (i2 >= 4 ? 8 : 0) + hi * 8;
                #pragma unroll
                for (int j2 = 0; j2 < 2; ++j2) {
                    int k = kbA + j2;
                    int c = k / 9; int rem = k - c * 9;
                    int r = rem / 3; int s = rem - r * 3;
                    a[2 * i2 + j2] = sIn[c * (HP * WP) + (oy + r) * WP + (ox + s)];
                }
                int kbB = kk + 2 * i2 + (i2 >= 4 ? 8 : 0) + hi * 8;
                b0[2 * i2]     = sU[kbB * 32 + l15];
                b0[2 * i2 + 1] = sU[(kbB + 1) * 32 + l15];
                b1[2 * i2]     = sU[kbB * 32 + 16 + l15];
                b1[2 * i2 + 1] = sU[(kbB + 1) * 32 + 16 + l15];
            }
            acc0 = __builtin_amdgcn_wmma_f32_16x16x32_f16(false, a, false, b0, (short)0, acc0, false, false);
            acc1 = __builtin_amdgcn_wmma_f32_16x16x32_f16(false, a, false, b1, (short)0, acc1, false, false);
        }
        float bv0 = bias[l15], bv1 = bias[16 + l15];
        _Float16* outE = outp + (size_t)e * 12800;
        #pragma unroll
        for (int r2 = 0; r2 < 8; ++r2) {
            int mo = mt * 16 + r2 + hi * 8;
            float v0 = acc0[r2] + bv0; v0 = v0 > 0.0f ? v0 : 0.01f * v0;
            float v1 = acc1[r2] + bv1; v1 = v1 > 0.0f ? v1 : 0.01f * v1;
            outE[(size_t)l15 * 400 + mo]        = (_Float16)v0;
            outE[(size_t)(16 + l15) * 400 + mo] = (_Float16)v1;
        }
    }
}

// ---------------- dense N-strip GEMM: out = relu(A @ W^T + b); one wave = 16 x (NT*16) ----------------
template<int NT>
__global__ __launch_bounds__(32)
void gemm_wmma_relu(const _Float16* __restrict__ A, const float* __restrict__ W,
                    const float* __restrict__ bias, _Float16* __restrict__ out, int K) {
    constexpr int N = NT * 16;
    int mt = blockIdx.x;
    int lane = threadIdx.x;
    int hi = lane >> 4, l15 = lane & 15;
    const _Float16* Arow = A + (size_t)(mt * 16 + l15) * K;
    v8f acc[NT];
    #pragma unroll
    for (int nt = 0; nt < NT; ++nt) acc[nt] = (v8f){};
    for (int kk = 0; kk < K; kk += 32) {
        __builtin_prefetch((const void*)(Arow + kk + 128), 0, 0);
        v16h a;
        #pragma unroll
        for (int i = 0; i < 8; ++i) {
            int kbA = kk + 2 * i + (i >= 4 ? 8 : 0) + hi * 8;
            a[2 * i]     = Arow[kbA];
            a[2 * i + 1] = Arow[kbA + 1];
        }
        #pragma unroll
        for (int nt = 0; nt < NT; ++nt) {
            const float* Wrow = W + (size_t)(nt * 16 + l15) * K;
            v16h bf;
            #pragma unroll
            for (int i = 0; i < 8; ++i) {
                int kbB = kk + 2 * i + (i >= 4 ? 8 : 0) + hi * 8;
                bf[2 * i]     = (_Float16)Wrow[kbB];
                bf[2 * i + 1] = (_Float16)Wrow[kbB + 1];
            }
            acc[nt] = __builtin_amdgcn_wmma_f32_16x16x32_f16(false, a, false, bf,
                                                             (short)0, acc[nt], false, false);
        }
    }
    #pragma unroll
    for (int nt = 0; nt < NT; ++nt) {
        int n = nt * 16 + l15;
        float bv = bias[n];
        #pragma unroll
        for (int r = 0; r < 8; ++r) {
            int m = mt * 16 + r + hi * 8;
            float v = acc[nt][r] + bv;
            v = v > 0.0f ? v : 0.0f;
            out[(size_t)m * N + n] = (_Float16)v;
        }
    }
}

// ---------------- final 64->1 heads, no relu ----------------
__global__ void heads_kernel(const _Float16* __restrict__ x2, const _Float16* __restrict__ e2,
                             const float* __restrict__ xw, const float* __restrict__ xb,
                             const float* __restrict__ ew, const float* __restrict__ eb,
                             float* __restrict__ out) {
    int e = blockIdx.x * blockDim.x + threadIdx.x;
    if (e >= NEV) return;
    float sx = xb[0], se = eb[0];
    #pragma unroll 8
    for (int k = 0; k < 64; ++k) {
        sx += (float)x2[(size_t)e * 64 + k] * xw[k];
        se += (float)e2[(size_t)e * 64 + k] * ew[k];
    }
    out[e * 2 + 0] = sx;
    out[e * 2 + 1] = se;
}

extern "C" void kernel_launch(void* const* d_in, const int* in_sizes, int n_in,
                              void* d_out, int out_size, void* d_ws, size_t ws_size,
                              hipStream_t stream) {
    const float* Traces  = (const float*)d_in[0];
    const float* Pstart  = (const float*)d_in[1];
    const float* l0_Wih  = (const float*)d_in[2];
    const float* l0_Whh  = (const float*)d_in[3];
    const float* l0_b    = (const float*)d_in[4];
    const float* l12_Wih = (const float*)d_in[5];
    const float* l12_Whh = (const float*)d_in[6];
    const float* l12_b   = (const float*)d_in[7];
    const float* cw0 = (const float*)d_in[8];  const float* cb0 = (const float*)d_in[9];
    const float* cw1 = (const float*)d_in[10]; const float* cb1 = (const float*)d_in[11];
    const float* cw2 = (const float*)d_in[12]; const float* cb2 = (const float*)d_in[13];
    const float* cw3 = (const float*)d_in[14]; const float* cb3 = (const float*)d_in[15];
    const float* D1_w = (const float*)d_in[16]; const float* D1_b = (const float*)d_in[17];
    const float* D2_w = (const float*)d_in[18]; const float* D2_b = (const float*)d_in[19];
    const float* D3_w = (const float*)d_in[20]; const float* D3_b = (const float*)d_in[21];
    const float* X1_w = (const float*)d_in[22]; const float* X1_b = (const float*)d_in[23];
    const float* X2_w = (const float*)d_in[24]; const float* X2_b = (const float*)d_in[25];
    const float* X3_w = (const float*)d_in[26]; const float* X3_b = (const float*)d_in[27];
    const float* E1_w = (const float*)d_in[28]; const float* E1_b = (const float*)d_in[29];
    const float* E2_w = (const float*)d_in[30]; const float* E2_b = (const float*)d_in[31];
    const float* E3_w = (const float*)d_in[32]; const float* E3_b = (const float*)d_in[33];
    const int* Xs = (const int*)d_in[34];
    const int* Ys = (const int*)d_in[35];
    const int* Ev = (const int*)d_in[36];

    const int P = P_TRACES;
    char* ws = (char*)d_ws;
    size_t off = 0;
    auto take = [&](size_t bytes) -> void* {
        void* p = ws + off;
        off = (off + bytes + 255) & ~(size_t)255;
        return p;
    };
    float*     pre  = (float*)take((size_t)P * 40 * 4);
    float*     seqA = (float*)take((size_t)P * 10 * 4);
    float*     seqB = (float*)take((size_t)P * 10 * 4);
    float*     Main = (float*)take((size_t)NEV * 2640 * 4);
    _Float16*  actA = (_Float16*)take((size_t)NEV * 12800 * 2);
    _Float16*  actB = (_Float16*)take((size_t)NEV * 12800 * 2);
    _Float16*  m1  = (_Float16*)take((size_t)NEV * 128 * 2);
    _Float16*  m2  = (_Float16*)take((size_t)NEV * 128 * 2);
    _Float16*  m3  = (_Float16*)take((size_t)NEV * 128 * 2);
    _Float16*  xh1 = (_Float16*)take((size_t)NEV * 128 * 2);
    _Float16*  xh2 = (_Float16*)take((size_t)NEV * 64 * 2);
    _Float16*  eh1 = (_Float16*)take((size_t)NEV * 128 * 2);
    _Float16*  eh2 = (_Float16*)take((size_t)NEV * 64 * 2);

    // 1) zero the scatter grid
    int nMain = NEV * 2640;
    zero_f32<<<(nMain + 255) / 256, 256, 0, stream>>>(Main, nMain);

    // 2) BiLSTM stack
    lstm_pre<<<(P * 40 + 255) / 256, 256, 0, stream>>>(Traces, 40, 40, l0_Wih, l0_b, pre, P);
    lstm_scan<<<2, 32, 0, stream>>>(pre, l0_Whh, seqA, P);
    lstm_pre<<<(P * 40 + 255) / 256, 256, 0, stream>>>(seqA, 10, 10, l12_Wih, l12_b, pre, P);
    lstm_scan<<<2, 32, 0, stream>>>(pre, l12_Whh, seqB, P);
    lstm_pre<<<(P * 40 + 255) / 256, 256, 0, stream>>>(seqB, 10, 10, l12_Wih + 400, l12_b + 40, pre, P);
    lstm_scan<<<2, 32, 0, stream>>>(pre, l12_Whh + 200, seqA, P);

    // 3) scatter into event grid
    scatter_grid<<<(P + 255) / 256, 256, 0, stream>>>(seqA, Pstart, Xs, Ys, Ev, Main, P);

    // 4) conv stack (WMMA implicit GEMM; conv1-3 input staged by TDM)
    conv0_lrelu<<<NEV, 128, 0, stream>>>(Main, cw0, cb0, actA);
    conv32_lrelu<<<NEV, 128, 0, stream>>>(actA, cw1, cb1, actB);
    conv32_lrelu<<<NEV, 128, 0, stream>>>(actB, cw2, cb2, actA);
    conv32_lrelu<<<NEV, 128, 0, stream>>>(actA, cw3, cb3, actB);

    // 5) dense stack (WMMA N-strips, A fragment reused across N tiles)
    const int MT = NEV / 16;  // 256 M-tiles
    gemm_wmma_relu<8><<<MT, 32, 0, stream>>>(actB, D1_w, D1_b, m1, 12800);
    gemm_wmma_relu<8><<<MT, 32, 0, stream>>>(m1,  D2_w, D2_b, m2, 128);
    gemm_wmma_relu<8><<<MT, 32, 0, stream>>>(m2,  D3_w, D3_b, m3, 128);
    gemm_wmma_relu<8><<<MT, 32, 0, stream>>>(m3,  X1_w, X1_b, xh1, 128);
    gemm_wmma_relu<4><<<MT, 32, 0, stream>>>(xh1, X2_w, X2_b, xh2, 128);
    gemm_wmma_relu<8><<<MT, 32, 0, stream>>>(m3,  E1_w, E1_b, eh1, 128);
    gemm_wmma_relu<4><<<MT, 32, 0, stream>>>(eh1, E2_w, E2_b, eh2, 128);

    // 6) final heads -> (NEV, 2)
    heads_kernel<<<(NEV + 255) / 256, 256, 0, stream>>>(xh2, eh2, X3_w, X3_b, E3_w, E3_b,
                                                        (float*)d_out);
}
